// SoftmaxAttention_9363028705743
// MI455X (gfx1250) — matrix operand order repaired and checked by reading, not compiled, not run on hardware
//
#include <hip/hip_runtime.h>

// Fused softmax-attention for MI455X (gfx1250, wave32, WMMA).
// Roofline: attn output = 402.7 MB fp32 stores -> ~19 us HBM floor @23.3TB/s.
// Matmuls via v_wmma_f32_16x16x32_f16 (f32 accum) -> compute sub-dominant.
// Softmax fully fp32. 32-row score tile resident in LDS (~257 KB of 320 KB).
// v2: b128 stats scans with 4 independent accumulators + branchless max;
//     1/sqrt(64) folded (bit-exactly, power of two) into the Q fragment.

typedef __attribute__((ext_vector_type(16))) _Float16 v16h;
typedef __attribute__((ext_vector_type(8)))  float    v8f;
typedef __attribute__((ext_vector_type(4)))  float    v4f;

#define B_   2
#define H_   12
#define N_   2048
#define D_   64
#define QT   32                 // query rows per workgroup
#define QB   (QT / 16)          // q-blocks (waves along q)  = 2
#define KQ   (8 / QB)           // waves along k             = 4
#define KCH  (N_ / KQ)          // k-columns per wave        = 512
#define TPR  (256 / QT)         // threads per row in stats  = 8
#define SEGW (N_ / TPR)         // cols per stats thread     = 256
#define SSTR (N_ + 4)           // LDS row stride (+4 floats: bank-conflict-free col gather)
#define NEG_BIG 1000000.0f

__device__ __forceinline__ float fmax_fin(float a, float b) {
  return (a > b) ? a : b;       // finite data only: no NaN canonicalization needed
}

__global__ __launch_bounds__(256, 1)
void attn_fused_kernel(const float* __restrict__ Q, const float* __restrict__ K,
                       const float* __restrict__ V, const float* __restrict__ Mk,
                       float* __restrict__ Out, float* __restrict__ Attn) {
  __shared__ float S[QT][SSTR];          // score tile, later p tile, later O-reduce scratch
  __shared__ float pmax[QT][TPR];
  __shared__ float psum[QT][TPR];
  __shared__ float mrow[QT];
  __shared__ float lrinv[QT];

  const int tid  = threadIdx.x;
  const int w    = tid >> 5;             // wave 0..7
  const int lane = tid & 31;
  const int m16  = lane & 15;
  const int kh   = lane >> 4;            // half-wave id
  const int qb   = w / KQ;               // q-block 0..1
  const int kq   = w % KQ;               // k-chunk 0..3

  const int bid = blockIdx.x;
  const int qt  = bid % (N_ / QT);
  const int bh  = bid / (N_ / QT);
  const int b   = bh / H_;
  const int q0  = qt * QT;

  // ---- Load Q A-fragments once (16x32 f16 A layout: lane=row m16, K-halves by kh).
  // Fold the 1/sqrt(64)=0.125 scale here: exact (power of two), saves per-tile FMAs.
  v16h aQ0, aQ1;
  {
    const float* qrow = Q + ((size_t)bh * N_ + q0 + qb * 16 + m16) * D_;
    #pragma unroll
    for (int i = 0; i < 16; ++i) {
      const int kk = kh * 16 + i;
      aQ0[i] = (_Float16)(qrow[kk] * 0.125f);        // d = 0..31
      aQ1[i] = (_Float16)(qrow[kk + 32] * 0.125f);   // d = 32..63
    }
  }

  // ---- Phase 1: S = (Q K^T)*0.125 - 1e6*(1-mask), written to LDS
  for (int t = 0; t < KCH / 16; ++t) {
    const int kc0 = kq * KCH + t * 16;
    // B 32x16 operand: lane=col n (=m16); this code's k order (same as A): half-wave kh holds K=kh*16+i
    const float* krow = K + ((size_t)bh * N_ + kc0 + m16) * D_ + kh * 16;
    v16h bK0, bK1;
    #pragma unroll
    for (int i = 0; i < 16; ++i) {
      bK0[i] = (_Float16)krow[i];         // d = kh*16 + i
      bK1[i] = (_Float16)krow[i + 32];    // d = 32 + kh*16 + i
    }
    v8f c = {};
    c = __builtin_amdgcn_wmma_f32_16x16x32_f16(false, aQ0, false, bK0, (short)0, c, false, false);
    c = __builtin_amdgcn_wmma_f32_16x16x32_f16(false, aQ1, false, bK1, (short)0, c, false, false);
    const float mb = -NEG_BIG * (1.0f - Mk[b * N_ + kc0 + m16]);
    // C layout: lanes 0-15 -> M=j, lanes 16-31 -> M=j+8; N = m16
    #pragma unroll
    for (int j = 0; j < 8; ++j) {
      const int mm = (lane < 16) ? j : (j + 8);
      S[qb * 16 + mm][kc0 + m16] = c[j] + mb;
    }
  }
  __syncthreads();

  // ---- Phase 2: fp32 row max and sum(exp) (stable softmax, matches reference)
  {
    const int r = tid / TPR;
    const int seg = tid % TPR;
    const float* srow = &S[r][seg * SEGW];

    float m0 = -1e30f, m1 = -1e30f, m2 = -1e30f, m3 = -1e30f;
    for (int c = 0; c < SEGW; c += 4) {
      v4f sv = *reinterpret_cast<const v4f*>(srow + c);   // ds_load_b128
      m0 = fmax_fin(m0, sv[0]);
      m1 = fmax_fin(m1, sv[1]);
      m2 = fmax_fin(m2, sv[2]);
      m3 = fmax_fin(m3, sv[3]);
    }
    pmax[r][seg] = fmax_fin(fmax_fin(m0, m1), fmax_fin(m2, m3));
    __syncthreads();
    if (tid < QT) {
      float mm = -1e30f;
      #pragma unroll
      for (int s2 = 0; s2 < TPR; ++s2) mm = fmax_fin(mm, pmax[tid][s2]);
      mrow[tid] = mm;
    }
    __syncthreads();
    const float mv = mrow[r];
    float s0 = 0.0f, s1 = 0.0f, s2 = 0.0f, s3 = 0.0f;
    for (int c = 0; c < SEGW; c += 4) {
      v4f sv = *reinterpret_cast<const v4f*>(srow + c);
      s0 += __expf(sv[0] - mv);
      s1 += __expf(sv[1] - mv);
      s2 += __expf(sv[2] - mv);
      s3 += __expf(sv[3] - mv);
    }
    psum[r][seg] = (s0 + s1) + (s2 + s3);
    __syncthreads();
    if (tid < QT) {
      float l = 0.0f;
      #pragma unroll
      for (int k2 = 0; k2 < TPR; ++k2) l += psum[tid][k2];
      lrinv[tid] = 1.0f / l;
    }
    __syncthreads();
  }

  // ---- Phase 3a: p = exp(s-m)/l; write attn (coalesced b128) and overwrite LDS
  {
    const int r0 = w * (QT / 8);
    for (int rr = 0; rr < QT / 8; ++rr) {
      const int r = r0 + rr;
      const float mv = mrow[r];
      const float li = lrinv[r];
      float* arow = Attn + ((size_t)bh * N_ + q0 + r) * (size_t)N_;
      for (int cb = 0; cb < N_ / 128; ++cb) {
        const int col = cb * 128 + lane * 4;
        v4f sv = *reinterpret_cast<v4f*>(&S[r][col]);
        v4f pv;
        #pragma unroll
        for (int e = 0; e < 4; ++e) pv[e] = __expf(sv[e] - mv) * li;
        *reinterpret_cast<v4f*>(&S[r][col]) = pv;
        *reinterpret_cast<v4f*>(arow + col) = pv;
      }
    }
  }
  __syncthreads();

  // ---- Phase 3b: partial O = P V over this wave's 512-k chunk (WMMA f16, f32 accum)
  v8f accO[4] = {};
  for (int ks = 0; ks < KCH / 32; ++ks) {
    const int kb = kq * KCH + ks * 32;
    v16h aP;
    #pragma unroll
    for (int i = 0; i < 16; ++i) {
      const int kk = kh * 16 + i;
      aP[i] = (_Float16)S[qb * 16 + m16][kb + kk];   // padded stride => no bank conflicts
    }
    #pragma unroll
    for (int dt = 0; dt < 4; ++dt) {
      const float* vp = V + ((size_t)bh * N_ + kb + kh * 16) * D_ + dt * 16 + m16;
      v16h bV;
      #pragma unroll
      for (int i = 0; i < 16; ++i) bV[i] = (_Float16)vp[(size_t)i * D_];
      accO[dt] = __builtin_amdgcn_wmma_f32_16x16x32_f16(false, aP, false, bV, (short)0, accO[dt], false, false);
    }
  }
  __syncthreads();

  // ---- Phase 4: reduce 4 k-chunk partials per q-block (reuse S as scratch), store out
  {
    float* red = &S[0][0];  // [QB][KQ][16][64] = 8192 floats
    #pragma unroll
    for (int dt = 0; dt < 4; ++dt) {
      #pragma unroll
      for (int j = 0; j < 8; ++j) {
        const int mm = (lane < 16) ? j : (j + 8);
        red[((qb * KQ + kq) * 16 + mm) * D_ + dt * 16 + m16] = accO[dt][j];
      }
    }
    __syncthreads();
    const int r2  = tid >> 3;         // 0..31
    const int d0  = (tid & 7) * 8;    // 0..56
    const int qb2 = r2 >> 4;
    const int mm2 = r2 & 15;
    float o[8];
    #pragma unroll
    for (int j = 0; j < 8; ++j) o[j] = 0.0f;
    #pragma unroll
    for (int k2 = 0; k2 < KQ; ++k2) {
      const int base = ((qb2 * KQ + k2) * 16 + mm2) * D_ + d0;
      #pragma unroll
      for (int j = 0; j < 8; ++j) o[j] += red[base + j];
    }
    float* op = Out + ((size_t)bh * N_ + q0 + r2) * D_ + d0;
    #pragma unroll
    for (int j = 0; j < 8; ++j) op[j] = o[j];
  }
}

extern "C" void kernel_launch(void* const* d_in, const int* in_sizes, int n_in,
                              void* d_out, int out_size, void* d_ws, size_t ws_size,
                              hipStream_t stream) {
  const float* Q  = (const float*)d_in[0];
  const float* K  = (const float*)d_in[1];
  const float* V  = (const float*)d_in[2];
  const float* Mk = (const float*)d_in[3];
  float* Out  = (float*)d_out;
  float* Attn = Out + (size_t)B_ * H_ * N_ * D_;   // tuple: (out, attn) concatenated

  dim3 grid(B_ * H_ * (N_ / QT));   // 2*12*64 = 1536 workgroups
  dim3 block(256);                  // 8 waves (wave32)
  attn_fused_kernel<<<grid, block, 0, stream>>>(Q, K, V, Mk, Out, Attn);

  (void)in_sizes; (void)n_in; (void)out_size; (void)d_ws; (void)ws_size;
}
